// CNNBlock_6691559047703
// MI455X (gfx1250) — matrix operand order repaired and checked by reading, not compiled, not run on hardware
//
#include <hip/hip_runtime.h>
#include <hip/hip_bf16.h>

// ---------------------------------------------------------------------------
// CNNBlock for MI455X (gfx1250): all matmuls via v_wmma_f32_16x16x32_bf16.
// Pipeline: init pads -> prep bf16 weights -> proj_in GEMM ->
//   2x[conv GEMM(8 images/block, async LDS weight staging)+stats ->
//      stats finalize -> BN+ReLU+residual] ->
//   proj_out GEMM + residual + LayerNorm.
// ---------------------------------------------------------------------------

typedef __attribute__((ext_vector_type(16))) __bf16        v16bf;
typedef __attribute__((ext_vector_type(8)))  float         v8f;
typedef __attribute__((ext_vector_type(4)))  unsigned int  u32x4;
typedef __attribute__((ext_vector_type(4)))  float         f32x4;
typedef __attribute__((ext_vector_type(4)))  unsigned short u16x4;
typedef __attribute__((__vector_size__(4 * sizeof(int)))) int vsi4;  // async-copy payload type

#define NB   4096      // batch
#define NS   120       // stabilizers
#define NH   256       // hidden
#define NC   128       // conv channels
#define NG   12        // grid side
#define NP   144       // NG*NG
#define CB   8         // conv: batch images per block (weight-staging amortization)

union Frag { v16bf v; u32x4 q[2]; unsigned short us[16]; };

__device__ __forceinline__ unsigned short f2bf(float f) {
  unsigned int u = __builtin_bit_cast(unsigned int, f);
  unsigned int r = u + 0x7FFFu + ((u >> 16) & 1u);
  return (unsigned short)(r >> 16);
}

__device__ __forceinline__ v8f wmma_bf16(const Frag& a, const Frag& b, v8f c) {
  return __builtin_amdgcn_wmma_f32_16x16x32_bf16(false, a.v, false, b.v,
                                                 (short)0, c, false, false);
}

// --------------------------------------------------------------------------
// K1: zero pad cells 120..143 of grid (f32) and act (bf16)
// --------------------------------------------------------------------------
__global__ void k_init(float* __restrict__ grid, unsigned short* __restrict__ act) {
  int idx = blockIdx.x * blockDim.x + threadIdx.x;
  const int npad = NB * 24 * NC;
  if (idx < npad) {
    int b   = idx / (24 * NC);
    int rem = idx - b * (24 * NC);
    int off = (b * NP + NS) * NC + rem;
    grid[off] = 0.0f;
    act[off]  = 0;
  }
}

// --------------------------------------------------------------------------
// K_prepw: one-time conv weight conversion f32 [L][O][I][3][3] ->
//          bf16 tap-major [L][t][o][i] (L2-resident, 576KB total)
// --------------------------------------------------------------------------
__global__ void k_prepw(const float* __restrict__ convw, unsigned short* __restrict__ wbf) {
  int i = blockIdx.x * blockDim.x + threadIdx.x;   // over 2*9*128*128
  if (i < 2 * 9 * NC * NC) {
    int L = i / (9 * NC * NC);
    int rem = i - L * 9 * NC * NC;
    int t = rem / (NC * NC);
    int rem2 = rem - t * NC * NC;
    int o = rem2 >> 7, ci = rem2 & 127;
    wbf[i] = f2bf(convw[(L * NC * NC + (o << 7) + ci) * 9 + t]);
  }
}

// --------------------------------------------------------------------------
// K0: proj_in  h = x @ W_in + b_in, scatter to grid[b][p][c] (f32 + bf16)
// 256 threads = 8 waves; wave -> 16 rows x 128 cols. K=256 (8 bf16 steps).
// dyn LDS: 64KB Wt (bf16 [128][256]) + 64KB store staging.
// --------------------------------------------------------------------------
__global__ void k_proj_in(const float* __restrict__ x, const float* __restrict__ Win,
                          const float* __restrict__ bin, float* __restrict__ grid,
                          unsigned short* __restrict__ act) {
  extern __shared__ char smem[];
  unsigned short* Wt = (unsigned short*)smem;                  // [128][256] bf16
  float* stage = (float*)(smem + NC * NH * 2);                 // 8 waves * 16*128 f32
  const int tid = threadIdx.x;
  for (int i = tid; i < NC * NH; i += blockDim.x) {
    int n = i >> 8, k = i & 255;
    Wt[i] = f2bf(Win[k * NC + n]);                             // transpose: Wt[n][k]
  }
  __syncthreads();

  const int wave = tid >> 5, lane = tid & 31;
  const int lhalf = lane >> 4, l16 = lane & 15;
  const int mbase = blockIdx.x * 128 + wave * 16;

  v8f acc[8];
#pragma unroll
  for (int n = 0; n < 8; ++n) {
    float bv = bin[n * 16 + l16];
#pragma unroll
    for (int r = 0; r < 8; ++r) acc[n][r] = bv;
  }

  const float* xrow = x + (mbase + l16) * NH;
#pragma unroll
  for (int ks = 0; ks < 8; ++ks) {
    const int k0 = ks * 32;
    Frag a;
    const float* p0 = xrow + k0 + lhalf * 8;
#pragma unroll
    for (int e = 0; e < 8; ++e) a.us[e] = f2bf(p0[e]);
#pragma unroll
    for (int e = 0; e < 8; ++e) a.us[8 + e] = f2bf(p0[16 + e]);
#pragma unroll
    for (int n = 0; n < 8; ++n) {
      Frag b;
      const unsigned short* wp = Wt + (n * 16 + l16) * NH + k0 + lhalf * 8;
      b.q[0] = *(const u32x4*)wp;
      b.q[1] = *(const u32x4*)(wp + 16);
      acc[n] = wmma_bf16(a, b, acc[n]);
    }
  }

  // dump accumulators to LDS tile, then store rows coalesced
  float* st = stage + wave * (16 * NC);
#pragma unroll
  for (int n = 0; n < 8; ++n)
#pragma unroll
    for (int r = 0; r < 8; ++r)
      st[(r + lhalf * 8) * NC + n * 16 + l16] = acc[n][r];
  __syncthreads();

#pragma unroll
  for (int r = 0; r < 16; ++r) {
    int row = mbase + r;
    int bb = row / NS, ss = row - bb * NS;
    int base = (bb * NP + ss) * NC + lane * 4;
    f32x4 v = *(f32x4*)&st[r * NC + lane * 4];
    *(f32x4*)&grid[base] = v;
    u16x4 h = { f2bf(v[0]), f2bf(v[1]), f2bf(v[2]), f2bf(v[3]) };
    *(u16x4*)&act[base] = h;
  }
}

// --------------------------------------------------------------------------
// K2: dilated 3x3 conv as 9-tap implicit GEMM + per-block BN partial stats.
// One block per CB batch images: 288 threads = 9 waves, wave w -> positions
// w*16..w*16+15, all 128 out channels. dyn LDS: 288KB tap-major W^T bf16,
// staged once per block via async global->LDS copy, reused for CB images.
// Stats reduced regs->shfl->LDS->partial buffer (deterministic, no atomics).
// --------------------------------------------------------------------------
__global__ void k_conv(const unsigned short* __restrict__ act,
                       const unsigned short* __restrict__ wbf,
                       const float* __restrict__ convb,
                       float* __restrict__ y, float* __restrict__ part, int dil) {
  extern __shared__ char smem[];
  unsigned short* Wt = (unsigned short*)smem;   // [9][o=128][i=128] bf16
  __shared__ float sS[9][NC];
  __shared__ float sQ[9][NC];
  const int tid = threadIdx.x;

  // ---- stage pre-converted bf16 weights into LDS --------------------------
#if defined(__gfx1250__) && __has_builtin(__builtin_amdgcn_global_load_async_to_lds_b128)
  for (int i = tid; i < 9 * NC * NC / 8; i += blockDim.x) {
    __builtin_amdgcn_global_load_async_to_lds_b128(
        (__attribute__((address_space(1))) vsi4*)(wbf + i * 8),
        (__attribute__((address_space(3))) vsi4*)(Wt + i * 8), 0, 0);
  }
#if __has_builtin(__builtin_amdgcn_s_wait_asynccnt)
  __builtin_amdgcn_s_wait_asynccnt(0);
#else
  asm volatile("s_wait_asynccnt 0" ::: "memory");
#endif
#else
  for (int i = tid; i < 9 * NC * NC / 8; i += blockDim.x)
    *(u32x4*)(Wt + i * 8) = *(const u32x4*)(wbf + i * 8);
#endif
  __syncthreads();

  const int wave = tid >> 5, lane = tid & 31;
  const int lhalf = lane >> 4, l16 = lane & 15;
  const int pbase = wave * 16;
  const int prow = pbase + l16;
  const int rr = prow / NG, cc = prow - rr * NG;
  const int bb0 = blockIdx.x * CB;

  float statS[8], statQ[8];
#pragma unroll
  for (int n = 0; n < 8; ++n) { statS[n] = 0.f; statQ[n] = 0.f; }

  for (int bi = 0; bi < CB; ++bi) {
    const int b = bb0 + bi;
    const unsigned short* actb = act + b * NP * NC;
    if (bi + 1 < CB) __builtin_prefetch(act + (b + 1) * NP * NC, 0, 1);

    v8f acc[8];
#pragma unroll
    for (int n = 0; n < 8; ++n) {
      float bv = convb[n * 16 + l16];
#pragma unroll
      for (int r = 0; r < 8; ++r) acc[n][r] = bv;
    }

    for (int t = 0; t < 9; ++t) {
      int dy = (t / 3 - 1) * dil, dx = (t % 3 - 1) * dil;
      int r2 = rr + dy, c2 = cc + dx;
      bool valid = (r2 >= 0) & (r2 < NG) & (c2 >= 0) & (c2 < NG);
      int psrc = valid ? (r2 * NG + c2) : prow;
      const unsigned short* arow = actb + psrc * NC;
#pragma unroll
      for (int ks = 0; ks < 4; ++ks) {
        Frag a;
        const unsigned short* p0 = arow + ks * 32 + lhalf * 8;
        a.q[0] = *(const u32x4*)p0;
        a.q[1] = *(const u32x4*)(p0 + 16);
        if (!valid) { a.q[0] = (u32x4)(0u); a.q[1] = (u32x4)(0u); }
#pragma unroll
        for (int n = 0; n < 8; ++n) {
          Frag bb;
          const unsigned short* wp = Wt + (t * NC + n * 16 + l16) * NC + ks * 32 + lhalf * 8;
          bb.q[0] = *(const u32x4*)wp;
          bb.q[1] = *(const u32x4*)(wp + 16);
          acc[n] = wmma_bf16(a, bb, acc[n]);
        }
      }
    }

    // store y[b][c][p] (each lane: 8 consecutive p) + accumulate stats in regs
    float* yb = y + b * NC * NP;
#pragma unroll
    for (int n = 0; n < 8; ++n) {
      int ch = n * 16 + l16;
#pragma unroll
      for (int r = 0; r < 8; ++r) { float v = acc[n][r]; statS[n] += v; statQ[n] += v * v; }
      int pb = pbase + lhalf * 8;
      f32x4 lo = { acc[n][0], acc[n][1], acc[n][2], acc[n][3] };
      f32x4 hi = { acc[n][4], acc[n][5], acc[n][6], acc[n][7] };
      *(f32x4*)&yb[ch * NP + pb]     = lo;
      *(f32x4*)&yb[ch * NP + pb + 4] = hi;
    }
  }

  // block-level deterministic stats reduction -> one partial record per block
#pragma unroll
  for (int n = 0; n < 8; ++n) {
    int ch = n * 16 + l16;
    float s0 = statS[n] + __shfl_xor(statS[n], 16);
    float q0 = statQ[n] + __shfl_xor(statQ[n], 16);
    if (lhalf == 0) { sS[wave][ch] = s0; sQ[wave][ch] = q0; }
  }
  __syncthreads();
  for (int c = tid; c < NC; c += blockDim.x) {
    float s = 0.f, q = 0.f;
    for (int w = 0; w < 9; ++w) { s += sS[w][c]; q += sQ[w][c]; }
    part[blockIdx.x * 256 + c]       = s;
    part[blockIdx.x * 256 + 128 + c] = q;
  }
}

// --------------------------------------------------------------------------
// K3: finalize BN stats -> per-channel scale/shift (deterministic reduce)
// --------------------------------------------------------------------------
__global__ void k_bnstats(const float* __restrict__ part, const float* __restrict__ gamma,
                          const float* __restrict__ beta, float* __restrict__ scale,
                          float* __restrict__ shift) {
  int c = threadIdx.x;
  float s = 0.f, q = 0.f;
  for (int b = 0; b < NB / CB; ++b) {
    s += part[b * 256 + c];
    q += part[b * 256 + 128 + c];
  }
  const float inv_n = 1.0f / ((float)NB * (float)NP);
  float mean = s * inv_n;
  float var  = q * inv_n - mean * mean;
  float sc   = rsqrtf(var + 1e-5f) * gamma[c];
  scale[c] = sc;
  shift[c] = beta[c] - mean * sc;
}

// --------------------------------------------------------------------------
// K4: grid += relu(scale*y + shift); act = bf16(grid). LDS transpose [c][p]->[p][c].
// --------------------------------------------------------------------------
__global__ void k_bnapply(const float* __restrict__ y, const float* __restrict__ scale,
                          const float* __restrict__ shift, float* __restrict__ grid,
                          unsigned short* __restrict__ act) {
  extern __shared__ char smem[];
  float* tile = (float*)smem;                 // [128][144]
  const int b = blockIdx.x, tid = threadIdx.x;
  const float* yb = y + b * NC * NP;
  for (int i = tid; i < NC * NP; i += blockDim.x) {
    int ch = i / NP;
    float v = yb[i] * scale[ch] + shift[ch];
    tile[i] = v > 0.f ? v : 0.f;
  }
  __syncthreads();
  float* gb = grid + b * NP * NC;
  unsigned short* ab = act + b * NP * NC;
  for (int i = tid; i < NP * NC; i += blockDim.x) {
    int p = i >> 7, ch = i & 127;
    float g = gb[i] + tile[ch * NP + p];
    gb[i] = g;
    ab[i] = f2bf(g);
  }
}

// --------------------------------------------------------------------------
// K5: proj_out GEMM (K=128, N=256) + residual(x) + LayerNorm(256).
// 256 threads = 8 waves; wave = (m-tile, col-half). dyn LDS 64KB reused:
// phase1 W^T bf16 [256][128], phase2 f32 dump [64][256].
// --------------------------------------------------------------------------
__global__ void k_proj_out(const unsigned short* __restrict__ act,
                           const float* __restrict__ Wout, const float* __restrict__ bout,
                           const float* __restrict__ x, const float* __restrict__ lng,
                           const float* __restrict__ lnb, float* __restrict__ out) {
  extern __shared__ char smem[];
  unsigned short* Wt = (unsigned short*)smem;   // [256][128] bf16 (phase 1)
  float* dump = (float*)smem;                   // [64][256] f32   (phase 2)
  const int tid = threadIdx.x;
  for (int i = tid; i < NH * NC; i += blockDim.x) {
    int n = i >> 7, k = i & 127;
    Wt[i] = f2bf(Wout[k * NH + n]);             // Wt[n][k]
  }
  __syncthreads();

  const int wave = tid >> 5, lane = tid & 31;
  const int lhalf = lane >> 4, l16 = lane & 15;
  const int mt = wave >> 1, nh = wave & 1;
  const int mbase = blockIdx.x * 64 + mt * 16;

  v8f acc[8];
#pragma unroll
  for (int n = 0; n < 8; ++n) {
    float bv = bout[nh * 128 + n * 16 + l16];
#pragma unroll
    for (int r = 0; r < 8; ++r) acc[n][r] = bv;
  }

  {
    int row = mbase + l16;
    int bb = row / NS, ss = row - bb * NS;
    const unsigned short* arow = act + (bb * NP + ss) * NC;
#pragma unroll
    for (int ks = 0; ks < 4; ++ks) {
      Frag a;
      const unsigned short* p0 = arow + ks * 32 + lhalf * 8;
      a.q[0] = *(const u32x4*)p0;
      a.q[1] = *(const u32x4*)(p0 + 16);
#pragma unroll
      for (int n = 0; n < 8; ++n) {
        Frag b;
        const unsigned short* wp = Wt + (nh * 128 + n * 16 + l16) * NC + ks * 32 + lhalf * 8;
        b.q[0] = *(const u32x4*)wp;
        b.q[1] = *(const u32x4*)(wp + 16);
        acc[n] = wmma_bf16(a, b, acc[n]);
      }
    }
  }
  __syncthreads();   // GEMM done everywhere; safe to repurpose LDS

#pragma unroll
  for (int n = 0; n < 8; ++n)
#pragma unroll
    for (int r = 0; r < 8; ++r)
      dump[(mt * 16 + r + lhalf * 8) * NH + nh * 128 + n * 16 + l16] = acc[n][r];
  __syncthreads();

  // LayerNorm: 64 rows / 8 waves = 8 rows each; lane owns 8 columns
  for (int rr = 0; rr < 8; ++rr) {
    int lrow = wave * 8 + rr;
    int grow = blockIdx.x * 64 + lrow;
    const float* xr = x + grow * NH;
    f32x4 d0 = *(f32x4*)&dump[lrow * NH + lane * 8];
    f32x4 d1 = *(f32x4*)&dump[lrow * NH + lane * 8 + 4];
    f32x4 x0 = *(const f32x4*)&xr[lane * 8];
    f32x4 x1 = *(const f32x4*)&xr[lane * 8 + 4];
    float v[8];
    float s = 0.f, q = 0.f;
#pragma unroll
    for (int e = 0; e < 4; ++e) {
      v[e] = d0[e] + x0[e]; s += v[e]; q += v[e] * v[e];
      v[4 + e] = d1[e] + x1[e]; s += v[4 + e]; q += v[4 + e] * v[4 + e];
    }
#pragma unroll
    for (int off = 16; off > 0; off >>= 1) {
      s += __shfl_xor(s, off);
      q += __shfl_xor(q, off);
    }
    float mu = s * (1.0f / NH);
    float var = q * (1.0f / NH) - mu * mu;
    float is = rsqrtf(var + 1e-5f);
    f32x4 g0 = *(const f32x4*)&lng[lane * 8];
    f32x4 g1 = *(const f32x4*)&lng[lane * 8 + 4];
    f32x4 b0 = *(const f32x4*)&lnb[lane * 8];
    f32x4 b1 = *(const f32x4*)&lnb[lane * 8 + 4];
    f32x4 o0, o1;
#pragma unroll
    for (int e = 0; e < 4; ++e) {
      o0[e] = (v[e] - mu) * is * g0[e] + b0[e];
      o1[e] = (v[4 + e] - mu) * is * g1[e] + b1[e];
    }
    float* orow = out + grow * NH;
    *(f32x4*)&orow[lane * 8]     = o0;
    *(f32x4*)&orow[lane * 8 + 4] = o1;
  }
}

// --------------------------------------------------------------------------
extern "C" void kernel_launch(void* const* d_in, const int* in_sizes, int n_in,
                              void* d_out, int out_size, void* d_ws, size_t ws_size,
                              hipStream_t stream) {
  (void)in_sizes; (void)n_in; (void)out_size; (void)ws_size;
  const float* x     = (const float*)d_in[0];
  const float* Win   = (const float*)d_in[1];
  const float* bin   = (const float*)d_in[2];
  const float* convw = (const float*)d_in[3];
  const float* convb = (const float*)d_in[4];
  const float* bng   = (const float*)d_in[5];
  const float* bnb   = (const float*)d_in[6];
  const float* Wout  = (const float*)d_in[7];
  const float* bout  = (const float*)d_in[8];
  const float* lng   = (const float*)d_in[9];
  const float* lnb   = (const float*)d_in[10];
  float* out = (float*)d_out;

  char* ws = (char*)d_ws;
  size_t offGrid = 0;
  size_t offAct  = offGrid + (size_t)NB * NP * NC * 4;   // grid f32
  size_t offY    = offAct  + (size_t)NB * NP * NC * 2;   // act bf16
  size_t offPart = offY    + (size_t)NB * NC * NP * 4;   // y f32
  size_t offStat = offPart + (size_t)NB * 256 * 4;       // partial stats
  size_t offWbf  = offStat + (size_t)2 * 256 * 4;        // scale/shift
  float* grid          = (float*)(ws + offGrid);
  unsigned short* act  = (unsigned short*)(ws + offAct);
  float* ybuf          = (float*)(ws + offY);
  float* part          = (float*)(ws + offPart);
  float* stat          = (float*)(ws + offStat);
  unsigned short* wbf  = (unsigned short*)(ws + offWbf); // [2][9][128][128] bf16

  k_init<<<(NB * 24 * NC + 255) / 256, 256, 0, stream>>>(grid, act);
  k_prepw<<<(2 * 9 * NC * NC + 255) / 256, 256, 0, stream>>>(convw, wbf);
  k_proj_in<<<(NB * NS) / 128, 256, 131072, stream>>>(x, Win, bin, grid, act);
  for (int L = 0; L < 2; ++L) {
    float* scale = stat + L * 256;
    float* shift = scale + 128;
    k_conv<<<NB / CB, 288, 9 * NC * NC * 2, stream>>>(act, wbf + (size_t)L * 9 * NC * NC,
                                                      convb + L * NC, ybuf, part,
                                                      L == 0 ? 1 : 2);
    k_bnstats<<<1, 128, 0, stream>>>(part, bng + L * NC, bnb + L * NC, scale, shift);
    k_bnapply<<<NB, 256, NC * NP * 4, stream>>>(ybuf, scale, shift, grid, act);
  }
  k_proj_out<<<(NB * NS) / 64, 256, 65536, stream>>>(act, Wout, bout, x, lng, lnb, out);
}
